// EmbedGCN_45286135169472
// MI455X (gfx1250) — matrix operand order repaired and checked by reading, not compiled, not run on hardware
//
#include <hip/hip_runtime.h>
#include <hip/hip_bf16.h>

typedef __attribute__((ext_vector_type(16))) __bf16 v16bf;
typedef __attribute__((ext_vector_type(8)))  float  v8f;

#define N_NODES 4000
#define D_IN    128
#define D_OUT   128
#define E_DIM   16
#define BATCH   8
#define NCOL    (BATCH * D_IN)   // 1024
#define LDSK    40               // 32 k + 8 pad (16 B) to break bank conflicts

// ---------------------------------------------------------------------------
// Kernel 1: A[n, :] = softmax(relu(E1[n] . E2[m]^T)) , stored bf16.
// One block per row; E2 (256 KB) stays resident in L2 across all 4000 blocks.
// ---------------------------------------------------------------------------
__global__ __launch_bounds__(256) void adj_softmax_kernel(
    const float* __restrict__ E1, const float* __restrict__ E2,
    __bf16* __restrict__ A)
{
    __shared__ float srow[N_NODES];
    __shared__ float sred[256];
    const int n = blockIdx.x;
    const int t = threadIdx.x;

    float e1[E_DIM];
    #pragma unroll
    for (int d = 0; d < E_DIM; ++d) e1[d] = E1[n * E_DIM + d];

    float lmax = 0.0f;                       // relu outputs are >= 0
    for (int m = t; m < N_NODES; m += 256) {
        const float4* p = reinterpret_cast<const float4*>(E2 + m * E_DIM);
        float acc = 0.0f;
        #pragma unroll
        for (int q = 0; q < E_DIM / 4; ++q) {
            float4 v = p[q];
            acc += e1[4*q+0]*v.x + e1[4*q+1]*v.y + e1[4*q+2]*v.z + e1[4*q+3]*v.w;
        }
        acc = fmaxf(acc, 0.0f);
        srow[m] = acc;
        lmax = fmaxf(lmax, acc);
    }
    sred[t] = lmax;
    __syncthreads();
    for (int s = 128; s > 0; s >>= 1) {
        if (t < s) sred[t] = fmaxf(sred[t], sred[t + s]);
        __syncthreads();
    }
    const float rmax = sred[0];
    __syncthreads();

    float lsum = 0.0f;
    for (int m = t; m < N_NODES; m += 256) {
        float e = __expf(srow[m] - rmax);
        srow[m] = e;
        lsum += e;
    }
    sred[t] = lsum;
    __syncthreads();
    for (int s = 128; s > 0; s >>= 1) {
        if (t < s) sred[t] += sred[t + s];
        __syncthreads();
    }
    const float rinv = 1.0f / sred[0];
    __syncthreads();

    for (int m = t; m < N_NODES; m += 256)
        A[(size_t)n * N_NODES + m] = (__bf16)(srow[m] * rinv);
}

// ---------------------------------------------------------------------------
// Pack x as XpT[c][m] (bf16), c = b*128 + f, m = node. LDS-tiled 32x32
// transpose so both the x reads and XpT writes are coalesced.
// ---------------------------------------------------------------------------
__global__ __launch_bounds__(256) void pack_xT_kernel(
    const float* __restrict__ x, __bf16* __restrict__ XpT)
{
    __shared__ float sT[32][33];
    const int c0 = blockIdx.x * 32;          // column block (b, f range)
    const int m0 = blockIdx.y * 32;          // node block
    const int b  = c0 >> 7;
    const int f0 = c0 & 127;
    const int tx = threadIdx.x;              // 0..31
    const int ty = threadIdx.y;              // 0..7

    #pragma unroll
    for (int j = 0; j < 4; ++j) {
        const int row = ty + j * 8;          // 0..31
        sT[row][tx] = x[((size_t)b * N_NODES + (m0 + row)) * D_IN + f0 + tx];
    }
    __syncthreads();
    #pragma unroll
    for (int j = 0; j < 4; ++j) {
        const int row = ty + j * 8;          // local c index
        XpT[(size_t)(c0 + row) * N_NODES + m0 + tx] = (__bf16)sT[tx][row];
    }
}

// WbT[o][f] = W[f][o], bf16  (tiny: 128x128)
__global__ __launch_bounds__(256) void pack_wT_kernel(
    const float* __restrict__ W, __bf16* __restrict__ WbT)
{
    int i = blockIdx.x * 256 + threadIdx.x;
    if (i >= D_IN * D_OUT) return;
    int o = i >> 7;
    int f = i & 127;
    WbT[i] = (__bf16)W[(size_t)f * D_OUT + o];
}

// ---------------------------------------------------------------------------
// Tiled bf16 WMMA GEMM: C[M,N] = A[M,K] @ B[K,N] with B given TRANSPOSED
// as Bt[N][K] row-major. f32 accumulate.
// Block tile 128x128, 8 waves arranged 2(M) x 4(N), each wave 64x32
// = 4x2 v_wmma_f32_16x16x32_bf16 per K-step of 32. Double-buffered LDS.
// All fragment loads hoisted ahead of the WMMA block so the 12 ds_load_b128
// issue before the first s_wait_dscnt and the 8 WMMAs run back-to-back.
// MODE 0: store bf16 C (row stride N).
// MODE 1: fuse bias + relu; row r = n*8 + b, scatter to out[b, n, col] f32.
// ---------------------------------------------------------------------------
template <int MODE>
__global__ __launch_bounds__(256) void wmma_gemm_bf16(
    const __bf16* __restrict__ Am,   // [M][K]
    const __bf16* __restrict__ Bt,   // [N][K]  (B transposed)
    int M, int N, int K,
    __bf16* __restrict__ Cbf,
    const float* __restrict__ bias,
    float* __restrict__ out)
{
    __shared__ __align__(16) __bf16 sA[2][128][LDSK];   // [m][k]
    __shared__ __align__(16) __bf16 sB[2][128][LDSK];   // [n][k]

    const int t     = threadIdx.x;
    const int lane  = t & 31;
    const int wave  = t >> 5;
    const int waveM = wave >> 2;          // 0..1
    const int waveN = wave & 3;           // 0..3
    const int half  = lane >> 4;          // lane half (WMMA layout)
    const int lidx  = lane & 15;
    const int m0    = blockIdx.x * 128;
    const int n0    = blockIdx.y * 128;

    v8f acc[4][2] = {};

    auto load_tiles = [&](int kt, int buf) {
        const int k0 = kt * 32;
        const int r  = t >> 1;               // 0..127
        const int c  = (t & 1) * 16;         // 0 or 16
        {   // A tile 128x32
            const int gr = m0 + r;
            uint4 v0 = make_uint4(0u, 0u, 0u, 0u);
            uint4 v1 = make_uint4(0u, 0u, 0u, 0u);
            if (gr < M) {
                const uint4* p =
                    reinterpret_cast<const uint4*>(Am + (size_t)gr * K + k0 + c);
                v0 = p[0];
                v1 = p[1];
                __builtin_prefetch(reinterpret_cast<const char*>(p) + 64, 0, 1);
            }
            *reinterpret_cast<uint4*>(&sA[buf][r][c])     = v0;
            *reinterpret_cast<uint4*>(&sA[buf][r][c + 8]) = v1;
        }
        {   // B tile (transposed source): rows are output columns
            const uint4* p =
                reinterpret_cast<const uint4*>(Bt + (size_t)(n0 + r) * K + k0 + c);
            uint4 v0 = p[0];
            uint4 v1 = p[1];
            __builtin_prefetch(reinterpret_cast<const char*>(p) + 64, 0, 1);
            *reinterpret_cast<uint4*>(&sB[buf][r][c])     = v0;
            *reinterpret_cast<uint4*>(&sB[buf][r][c + 8]) = v1;
        }
    };

    union Frag { uint4 q[2]; v16bf v; };

    const int nk = K / 32;
    load_tiles(0, 0);
    __syncthreads();

    for (int kt = 0; kt < nk; ++kt) {
        const int cur = kt & 1;
        if (kt + 1 < nk) load_tiles(kt + 1, cur ^ 1);

        // ---- Hoist ALL fragment loads (12x ds_load_b128) before the WMMAs.
        // B fragments (32x16): lane = column, element e -> k = 16*half + e.
        v16bf bfrag[2];
        #pragma unroll
        for (int nt = 0; nt < 2; ++nt) {
            const int nn = waveN * 32 + nt * 16 + lidx;
            Frag fb;
            fb.q[0] = *reinterpret_cast<const uint4*>(&sB[cur][nn][half * 16]);
            fb.q[1] = *reinterpret_cast<const uint4*>(&sB[cur][nn][half * 16 + 8]);
            bfrag[nt] = fb.v;
        }
        // A fragments (16x32): e -> k = 16*half + e
        // => two contiguous uint4 reads at k = 16*half and 16*half + 8.
        v16bf afrag[4];
        #pragma unroll
        for (int mt = 0; mt < 4; ++mt) {
            const int mm = waveM * 64 + mt * 16 + lidx;
            Frag fa;
            fa.q[0] = *reinterpret_cast<const uint4*>(&sA[cur][mm][half * 16]);
            fa.q[1] = *reinterpret_cast<const uint4*>(&sA[cur][mm][half * 16 + 8]);
            afrag[mt] = fa.v;
        }

        // ---- 8 back-to-back WMMAs on the matrix pipe.
        #pragma unroll
        for (int mt = 0; mt < 4; ++mt) {
            #pragma unroll
            for (int nt = 0; nt < 2; ++nt)
                acc[mt][nt] = __builtin_amdgcn_wmma_f32_16x16x32_bf16(
                    false, afrag[mt], false, bfrag[nt],
                    (short)0, acc[mt][nt], false, false);
        }
        __syncthreads();
    }

    // Epilogue. C/D layout: vgpr v -> m = v + 8*half, lane&15 = n.
    #pragma unroll
    for (int mt = 0; mt < 4; ++mt) {
        #pragma unroll
        for (int nt = 0; nt < 2; ++nt) {
            const int col = n0 + waveN * 32 + nt * 16 + lidx;
            #pragma unroll
            for (int v = 0; v < 8; ++v) {
                const int row = m0 + waveM * 64 + mt * 16 + half * 8 + v;
                if (row < M) {
                    float val = acc[mt][nt][v];
                    if (MODE == 0) {
                        Cbf[(size_t)row * N + col] = (__bf16)val;
                    } else {
                        val = fmaxf(val + bias[col], 0.0f);
                        const int bb = row & 7;     // row = n*8 + b
                        const int nn = row >> 3;
                        out[((size_t)bb * N_NODES + nn) * D_OUT + col] = val;
                    }
                }
            }
        }
    }
}

// ---------------------------------------------------------------------------
extern "C" void kernel_launch(void* const* d_in, const int* in_sizes, int n_in,
                              void* d_out, int out_size, void* d_ws, size_t ws_size,
                              hipStream_t stream) {
    const float* x    = (const float*)d_in[0];
    const float* E1   = (const float*)d_in[1];
    const float* E2   = (const float*)d_in[2];
    const float* W    = (const float*)d_in[3];
    const float* bias = (const float*)d_in[4];
    float* out = (float*)d_out;

    char* ws = (char*)d_ws;
    __bf16* A   = (__bf16*)(ws);                  // 4000*4000*2 = 32,000,000 B
    __bf16* XpT = (__bf16*)(ws + 32000000);       // 1024*4000*2 =  8,192,000 B
    __bf16* Agg = (__bf16*)(ws + 40192000);       // 32000*128*2 =  8,192,000 B
    __bf16* WbT = (__bf16*)(ws + 48384000);       //   128*128*2 =     32,768 B

    // Stage 1: adjacency softmax -> bf16
    adj_softmax_kernel<<<N_NODES, 256, 0, stream>>>(E1, E2, A);

    // Pack operands (B operands stored transposed: [outCol][k])
    dim3 gpx(NCOL / 32, N_NODES / 32);
    pack_xT_kernel<<<gpx, dim3(32, 8), 0, stream>>>(x, XpT);
    pack_wT_kernel<<<(D_IN * D_OUT + 255) / 256, 256, 0, stream>>>(W, WbT);

    // Stage 2: Agg[N, B*F] = A @ Xp   (M=4000, K=4000, N=1024)
    dim3 g2((N_NODES + 127) / 128, NCOL / 128);
    wmma_gemm_bf16<0><<<g2, 256, 0, stream>>>(
        A, XpT, N_NODES, NCOL, N_NODES, Agg, nullptr, nullptr);

    // Stage 3: out = relu(Agg @ W + b)  (M=32000, K=128, N=128), fused epilogue
    dim3 g3((N_NODES * BATCH) / 128, D_OUT / 128);
    wmma_gemm_bf16<1><<<g3, 256, 0, stream>>>(
        Agg, WbT, N_NODES * BATCH, D_OUT, D_IN, nullptr, bias, out);
}
